// PerceiverAttention_12592844112056
// MI455X (gfx1250) — matrix operand order repaired and checked by reading, not compiled, not run on hardware
//
#include <hip/hip_runtime.h>
#include <hip/hip_bf16.h>
#include <stdint.h>

typedef __bf16 bf16_t;
typedef __attribute__((ext_vector_type(16))) __bf16 v16bf;
typedef __attribute__((ext_vector_type(8)))  __bf16 v8bf;
typedef __attribute__((ext_vector_type(8)))  float  v8f;

#define DIMX     1024
#define HEADS    8
#define DIM_HEAD 64
#define INNERX   512
#define NSEQ     4096
#define MLAT     64
#define BATCH    8
#define NTOT     (NSEQ + MLAT)      /* 4160 */
#define ROWS_TOT (BATCH * NTOT)     /* 33280 */
#define NCHUNK   13                 /* j-dimension split for attention */
#define TILES_PER_CHUNK 5           /* 13 * 5 * 64 = 4160 */

// ---------------------------------------------------------------------------
// WMMA bf16 fragment loaders (wave32, 16x16x32 shapes).
// A 16x32: lane<16 holds K{0..7,16..23}, lane>=16 holds K{8..15,24..31}.
// B 32x16 (loaded from row-major B^T [N,K]): lane<16 K{0..7,16..23}, lane>=16 K{8..15,24..31}.
// ---------------------------------------------------------------------------
__device__ __forceinline__ v16bf load_frag_a(const bf16_t* row_base, int kb, int lane) {
  const bf16_t* p = row_base + kb + ((lane & 16) ? 8 : 0);
  union { v16bf v; v8bf h[2]; } u;
  u.h[0] = *(const v8bf*)(p);
  u.h[1] = *(const v8bf*)(p + 16);
  return u.v;
}
__device__ __forceinline__ v16bf load_frag_b(const bf16_t* row_base, int kb, int lane) {
  const bf16_t* p = row_base + kb + ((lane & 16) ? 8 : 0);
  union { v16bf v; v8bf h[2]; } u;
  u.h[0] = *(const v8bf*)(p);
  u.h[1] = *(const v8bf*)(p + 16);
  return u.v;
}

// ---------------------------------------------------------------------------
// Weight transpose + cast: dst[N][K] = (bf16)src[K][N]
// ---------------------------------------------------------------------------
__global__ void transpose_w_kernel(const float* __restrict__ src, bf16_t* __restrict__ dst,
                                   int K, int N) {
  size_t idx = (size_t)blockIdx.x * blockDim.x + threadIdx.x;
  if (idx >= (size_t)K * N) return;
  int k = (int)(idx % K);
  int n = (int)(idx / K);
  dst[idx] = (bf16_t)src[(size_t)k * N + n];
}

// ---------------------------------------------------------------------------
// LayerNorm of x and latents -> bf16 kv_in [8, 4160, 1024]
// ---------------------------------------------------------------------------
__global__ __launch_bounds__(256)
void ln_input_kernel(const float* __restrict__ x, const float* __restrict__ lat,
                     const float* __restrict__ gx, const float* __restrict__ bx,
                     const float* __restrict__ gl, const float* __restrict__ bl,
                     bf16_t* __restrict__ kv_in) {
  __shared__ float s1[256], s2[256];
  const int row = blockIdx.x;
  const int bb = row / NTOT, rr = row % NTOT;
  const float* src; const float* g; const float* be;
  if (rr < NSEQ) { src = x  + ((size_t)bb * NSEQ + rr) * DIMX;           g = gx; be = bx; }
  else           { src = lat + ((size_t)bb * MLAT + (rr - NSEQ)) * DIMX; g = gl; be = bl; }
  const int tid = threadIdx.x;
  float v[4]; float sum = 0.f, sq = 0.f;
#pragma unroll
  for (int i = 0; i < 4; ++i) { float t = src[tid + i * 256]; v[i] = t; sum += t; sq += t * t; }
  s1[tid] = sum; s2[tid] = sq; __syncthreads();
  for (int off = 128; off > 0; off >>= 1) {
    if (tid < off) { s1[tid] += s1[tid + off]; s2[tid] += s2[tid + off]; }
    __syncthreads();
  }
  float mu   = s1[0] * (1.f / DIMX);
  float var  = s2[0] * (1.f / DIMX) - mu * mu;
  float rstd = rsqrtf(var + 1e-5f);
  bf16_t* drow = kv_in + (size_t)row * DIMX;
#pragma unroll
  for (int i = 0; i < 4; ++i) {
    int c = tid + i * 256;
    drow[c] = (bf16_t)((v[i] - mu) * rstd * g[c] + be[c]);
  }
}

// ---------------------------------------------------------------------------
// Final LayerNorm: proj f32 [512,1024] -> d_out f32
// ---------------------------------------------------------------------------
__global__ __launch_bounds__(256)
void ln_out_kernel(const float* __restrict__ src, const float* __restrict__ g,
                   const float* __restrict__ be, float* __restrict__ dst) {
  __shared__ float s1[256], s2[256];
  const int row = blockIdx.x;
  const float* srow = src + (size_t)row * DIMX;
  const int tid = threadIdx.x;
  float v[4]; float sum = 0.f, sq = 0.f;
#pragma unroll
  for (int i = 0; i < 4; ++i) { float t = srow[tid + i * 256]; v[i] = t; sum += t; sq += t * t; }
  s1[tid] = sum; s2[tid] = sq; __syncthreads();
  for (int off = 128; off > 0; off >>= 1) {
    if (tid < off) { s1[tid] += s1[tid + off]; s2[tid] += s2[tid + off]; }
    __syncthreads();
  }
  float mu   = s1[0] * (1.f / DIMX);
  float var  = s2[0] * (1.f / DIMX) - mu * mu;
  float rstd = rsqrtf(var + 1e-5f);
  float* drow = dst + (size_t)row * DIMX;
#pragma unroll
  for (int i = 0; i < 4; ++i) {
    int c = tid + i * 256;
    drow[c] = (v[i] - mu) * rstd * g[c] + be[c];
  }
}

// ---------------------------------------------------------------------------
// 64x64-per-wave bf16 WMMA GEMM: C = A[M,K] @ B  (B given transposed: Bt[N,K]).
// grid = (N/64, Mtiles, batches), block = 32 (one wave).
// ---------------------------------------------------------------------------
__global__ __launch_bounds__(32)
void gemm64_bf16(const bf16_t* __restrict__ A, size_t a_bstride, int lda,
                 const bf16_t* __restrict__ Bt,
                 void* __restrict__ Cout, size_t c_bstride, int ldc,
                 int K, int f32out) {
  const int lane = threadIdx.x;
  const int m0 = blockIdx.y * 64;
  const int n0 = blockIdx.x * 64;
  const bf16_t* Ab = A + (size_t)blockIdx.z * a_bstride;
  const int r = lane & 15;

  v8f acc[4][4];
#pragma unroll
  for (int mi = 0; mi < 4; ++mi)
#pragma unroll
    for (int nj = 0; nj < 4; ++nj)
#pragma unroll
      for (int t = 0; t < 8; ++t) acc[mi][nj][t] = 0.f;

  for (int kb = 0; kb < K; kb += 32) {
    v16bf a[4], b[4];
#pragma unroll
    for (int mi = 0; mi < 4; ++mi)
      a[mi] = load_frag_a(Ab + (size_t)(m0 + mi * 16 + r) * lda, kb, lane);
#pragma unroll
    for (int nj = 0; nj < 4; ++nj)
      b[nj] = load_frag_b(Bt + (size_t)(n0 + nj * 16 + r) * K, kb, lane);
    if (kb + 32 < K)
      __builtin_prefetch(Ab + (size_t)(m0 + r) * lda + kb + 32, 0, 1);
#pragma unroll
    for (int mi = 0; mi < 4; ++mi)
#pragma unroll
      for (int nj = 0; nj < 4; ++nj)
        acc[mi][nj] = __builtin_amdgcn_wmma_f32_16x16x32_bf16(
            false, a[mi], false, b[nj], (short)0, acc[mi][nj], false, false);
  }

  const int half = lane >> 4, nf = lane & 15;
#pragma unroll
  for (int mi = 0; mi < 4; ++mi)
#pragma unroll
    for (int nj = 0; nj < 4; ++nj)
#pragma unroll
      for (int t = 0; t < 8; ++t) {
        size_t row = (size_t)(m0 + mi * 16 + t + 8 * half);
        size_t col = (size_t)(n0 + nj * 16 + nf);
        size_t idx = (size_t)blockIdx.z * c_bstride + row * (size_t)ldc + col;
        float vv = acc[mi][nj][t];
        if (f32out) ((float*)Cout)[idx] = vv;
        else        ((bf16_t*)Cout)[idx] = (bf16_t)vv;
      }
}

// ---------------------------------------------------------------------------
// V transpose: vT[b,h,d, j] = kv[b, j, 512 + h*64 + d]   (bf16, LDS tiled)
// grid = (NTOT/64, BATCH*HEADS), block = 256
// ---------------------------------------------------------------------------
__global__ __launch_bounds__(256)
void transpose_v_kernel(const bf16_t* __restrict__ kv, bf16_t* __restrict__ vT) {
  __shared__ __align__(16) bf16_t tile[64][65];
  const int jt = blockIdx.x;
  const int bh = blockIdx.y;
  const int bb = bh >> 3, h = bh & 7;
  const int tid = threadIdx.x;
  {
    int rr = tid >> 2;
    int d0 = (tid & 3) * 16;
    const bf16_t* src = kv + ((size_t)(bb * NTOT + jt * 64 + rr)) * DIMX + INNERX + h * DIM_HEAD + d0;
    v8bf u0 = *(const v8bf*)(src);
    v8bf u1 = *(const v8bf*)(src + 8);
#pragma unroll
    for (int i = 0; i < 8; ++i) tile[rr][d0 + i] = u0[i];
#pragma unroll
    for (int i = 0; i < 8; ++i) tile[rr][d0 + 8 + i] = u1[i];
  }
  __syncthreads();
  {
    int d   = tid >> 2;
    int jj0 = (tid & 3) * 16;
    bf16_t* dst = vT + ((size_t)(bh * 64 + d)) * NTOT + jt * 64 + jj0;
    union { v8bf v; bf16_t e[8]; } o0, o1;
#pragma unroll
    for (int i = 0; i < 8; ++i) { o0.e[i] = tile[jj0 + i][d]; o1.e[i] = tile[jj0 + 8 + i][d]; }
    *(v8bf*)(dst)     = o0.v;
    *(v8bf*)(dst + 8) = o1.v;
  }
}

// ---------------------------------------------------------------------------
// Flash attention, split along j into NCHUNK chunks for occupancy.
// grid = (NCHUNK, BATCH*HEADS); block = 128 (4 waves x 16 query rows).
// Emits unnormalized partials: pout f32 [bh][c][64][64], pm/pl f32 [bh][c][64].
// ---------------------------------------------------------------------------
__global__ __launch_bounds__(128)
void attn_partial_kernel(const bf16_t* __restrict__ q, const bf16_t* __restrict__ kv,
                         const bf16_t* __restrict__ vT, const int* __restrict__ mask,
                         float* __restrict__ pout, float* __restrict__ pm,
                         float* __restrict__ pl) {
  __shared__ __align__(16) bf16_t plds[4][16 * 72];
  const int wave = threadIdx.x >> 5, lane = threadIdx.x & 31;
  const int c = blockIdx.x;
  const int bh = blockIdx.y;
  const int b = bh >> 3, h = bh & 7;
  const int i0 = wave * 16;
  const int r = lane & 15, half = lane >> 4;
  const float scale2 = 0.015625f;   // (dh^-0.5)^2, scale applied twice per reference

  v16bf aq[2];
  const bf16_t* qrow = q + (size_t)(b * 64 + i0 + r) * INNERX + h * DIM_HEAD;
#pragma unroll
  for (int ks = 0; ks < 2; ++ks) aq[ks] = load_frag_a(qrow, ks * 32, lane);

  float m[8], l[8];
  v8f acc[4];
#pragma unroll
  for (int t = 0; t < 8; ++t) { m[t] = -1e30f; l[t] = 0.f; }
#pragma unroll
  for (int nd = 0; nd < 4; ++nd)
#pragma unroll
    for (int t = 0; t < 8; ++t) acc[nd][t] = 0.f;

  const bf16_t* kbase = kv + (size_t)b * NTOT * DIMX + h * DIM_HEAD;
  const bf16_t* vbase = vT + (size_t)(b * HEADS + h) * DIM_HEAD * NTOT;
  const int* mrow = mask + b * NSEQ;

  const int jt_begin = c * TILES_PER_CHUNK;
  for (int jt = jt_begin; jt < jt_begin + TILES_PER_CHUNK; ++jt) {
    const int j0 = jt * 64;
    v8f s[4];
#pragma unroll
    for (int nj = 0; nj < 4; ++nj)
#pragma unroll
      for (int t = 0; t < 8; ++t) s[nj][t] = 0.f;

    // sim tile = q @ k^T
#pragma unroll
    for (int ks = 0; ks < 2; ++ks) {
#pragma unroll
      for (int nj = 0; nj < 4; ++nj) {
        v16bf bk = load_frag_b(kbase + (size_t)(j0 + nj * 16 + r) * DIMX, ks * 32, lane);
        s[nj] = __builtin_amdgcn_wmma_f32_16x16x32_bf16(
            false, aq[ks], false, bk, (short)0, s[nj], false, false);
      }
    }
    // scale twice + mask (latent columns j>=NSEQ always kept)
#pragma unroll
    for (int nj = 0; nj < 4; ++nj) {
      int j = j0 + nj * 16 + r;
      bool keep = (j >= NSEQ) || (mrow[j] != 0);
#pragma unroll
      for (int t = 0; t < 8; ++t) {
        float sv = s[nj][t] * scale2;
        s[nj][t] = keep ? sv : -1e30f;
      }
    }
    // online softmax; row stats live per (VGPR t, lane-half) matching C layout
    float alpha[8], rs[8];
#pragma unroll
    for (int t = 0; t < 8; ++t) {
      float tm = fmaxf(fmaxf(s[0][t], s[1][t]), fmaxf(s[2][t], s[3][t]));
      tm = fmaxf(tm, __shfl_xor(tm, 1, 16));
      tm = fmaxf(tm, __shfl_xor(tm, 2, 16));
      tm = fmaxf(tm, __shfl_xor(tm, 4, 16));
      tm = fmaxf(tm, __shfl_xor(tm, 8, 16));
      float mn = fmaxf(m[t], tm);
      alpha[t] = __expf(m[t] - mn);
      m[t] = mn; rs[t] = 0.f;
    }
#pragma unroll
    for (int nj = 0; nj < 4; ++nj)
#pragma unroll
      for (int t = 0; t < 8; ++t) {
        float p = __expf(s[nj][t] - m[t]);
        rs[t] += p;
        plds[wave][(t + 8 * half) * 72 + nj * 16 + r] = (bf16_t)p;
      }
#pragma unroll
    for (int t = 0; t < 8; ++t) {
      float sum = rs[t];
      sum += __shfl_xor(sum, 1, 16);
      sum += __shfl_xor(sum, 2, 16);
      sum += __shfl_xor(sum, 4, 16);
      sum += __shfl_xor(sum, 8, 16);
      l[t] = l[t] * alpha[t] + sum;
#pragma unroll
      for (int nd = 0; nd < 4; ++nd) acc[nd][t] *= alpha[t];
    }
    // out += P @ V  (P via LDS round-trip to A layout; V from vT, contiguous in j)
#pragma unroll
    for (int ks = 0; ks < 2; ++ks) {
      const bf16_t* prow = &plds[wave][r * 72];
      v16bf ap = load_frag_a(prow, ks * 32, lane);
#pragma unroll
      for (int nd = 0; nd < 4; ++nd) {
        v16bf bv = load_frag_b(vbase + (size_t)(nd * 16 + r) * NTOT, j0 + ks * 32, lane);
        acc[nd] = __builtin_amdgcn_wmma_f32_16x16x32_bf16(
            false, ap, false, bv, (short)0, acc[nd], false, false);
      }
    }
  }
  // write unnormalized partials + row stats
  const size_t pbase = ((size_t)bh * NCHUNK + c) * 64;
#pragma unroll
  for (int nd = 0; nd < 4; ++nd)
#pragma unroll
    for (int t = 0; t < 8; ++t) {
      size_t row = pbase + i0 + t + 8 * half;
      pout[row * 64 + nd * 16 + r] = acc[nd][t];
    }
#pragma unroll
  for (int t = 0; t < 8; ++t) {
    if (r == t) {
      size_t row = pbase + i0 + t + 8 * half;
      pm[row] = m[t];
      pl[row] = l[t];
    }
  }
}

// ---------------------------------------------------------------------------
// Log-sum-exp merge of NCHUNK partials -> aout bf16 [8*64, 512]
// grid = BATCH*HEADS, block = 256
// ---------------------------------------------------------------------------
__global__ __launch_bounds__(256)
void attn_combine_kernel(const float* __restrict__ pout, const float* __restrict__ pm,
                         const float* __restrict__ pl, bf16_t* __restrict__ aout) {
  __shared__ float wgt[NCHUNK][64];
  __shared__ float invL[64];
  const int bh = blockIdx.x;
  const int b = bh >> 3, h = bh & 7;
  const int tid = threadIdx.x;
  const size_t sbase = (size_t)bh * NCHUNK * 64;

  if (tid < 64) {
    float M = -3.0e38f;
#pragma unroll
    for (int c = 0; c < NCHUNK; ++c) M = fmaxf(M, pm[sbase + c * 64 + tid]);
    float L = 0.f;
#pragma unroll
    for (int c = 0; c < NCHUNK; ++c) {
      float w = __expf(pm[sbase + c * 64 + tid] - M);
      wgt[c][tid] = w;
      L += w * pl[sbase + c * 64 + tid];
    }
    invL[tid] = 1.f / L;
  }
  __syncthreads();

  const int i  = tid >> 2;
  const int d0 = (tid & 3) * 16;
  float o[16];
#pragma unroll
  for (int dd = 0; dd < 16; ++dd) o[dd] = 0.f;
#pragma unroll
  for (int c = 0; c < NCHUNK; ++c) {
    const float* src = pout + ((sbase + c * 64 + i) * 64 + d0);
    float w = wgt[c][i];
#pragma unroll
    for (int dd = 0; dd < 16; ++dd) o[dd] += w * src[dd];
  }
  float sc = invL[i];
  bf16_t* dst = aout + (size_t)(b * 64 + i) * INNERX + h * DIM_HEAD + d0;
#pragma unroll
  for (int dd = 0; dd < 16; ++dd) dst[dd] = (bf16_t)(o[dd] * sc);
}

// ---------------------------------------------------------------------------
extern "C" void kernel_launch(void* const* d_in, const int* in_sizes, int n_in,
                              void* d_out, int out_size, void* d_ws, size_t ws_size,
                              hipStream_t stream) {
  (void)in_sizes; (void)n_in; (void)out_size; (void)ws_size;
  const float* x   = (const float*)d_in[0];
  const float* lat = (const float*)d_in[1];
  const int*   msk = (const int*)d_in[2];
  const float* Wq  = (const float*)d_in[3];
  const float* Wkv = (const float*)d_in[4];
  const float* Wo  = (const float*)d_in[5];
  const float* gx  = (const float*)d_in[6];
  const float* bx  = (const float*)d_in[7];
  const float* gl  = (const float*)d_in[8];
  const float* bl  = (const float*)d_in[9];
  const float* go  = (const float*)d_in[10];
  const float* bo  = (const float*)d_in[11];
  float* out = (float*)d_out;

  uintptr_t p = (uintptr_t)d_ws;
  auto carve = [&p](size_t bytes) -> void* {
    void* r = (void*)p; p += (bytes + 255) & ~(size_t)255; return r;
  };
  bf16_t* kv_in = (bf16_t*)carve((size_t)ROWS_TOT * DIMX * 2);                 // 68 MB
  bf16_t* kvout = (bf16_t*)carve((size_t)ROWS_TOT * DIMX * 2);                 // 68 MB
  bf16_t* WqT   = (bf16_t*)carve((size_t)INNERX * DIMX * 2);
  bf16_t* WkvT  = (bf16_t*)carve((size_t)DIMX * DIMX * 2);
  bf16_t* WoT   = (bf16_t*)carve((size_t)DIMX * INNERX * 2);
  bf16_t* qb    = (bf16_t*)carve((size_t)BATCH * MLAT * INNERX * 2);
  bf16_t* aout  = (bf16_t*)carve((size_t)BATCH * MLAT * INNERX * 2);
  float*  proj  = (float*)carve((size_t)BATCH * MLAT * DIMX * 4);
  float*  pout  = (float*)carve((size_t)BATCH * HEADS * NCHUNK * 64 * 64 * 4); // 13.6 MB
  float*  pm    = (float*)carve((size_t)BATCH * HEADS * NCHUNK * 64 * 4);
  float*  pl    = (float*)carve((size_t)BATCH * HEADS * NCHUNK * 64 * 4);
  // vT [b,h,64,4160] bf16 aliases kv_in: kv_in's last read (q GEMM) is launched
  // before transpose_v writes vT; stream order makes the reuse safe.
  bf16_t* vT    = kv_in;

  // 1) weights -> transposed bf16
  transpose_w_kernel<<<(DIMX * INNERX + 255) / 256, 256, 0, stream>>>(Wq,  WqT,  DIMX,  INNERX);
  transpose_w_kernel<<<(DIMX * DIMX   + 255) / 256, 256, 0, stream>>>(Wkv, WkvT, DIMX,  DIMX);
  transpose_w_kernel<<<(INNERX * DIMX + 255) / 256, 256, 0, stream>>>(Wo,  WoT,  INNERX, DIMX);

  // 2) layernorm x & latents -> bf16 kv_in [8,4160,1024]
  ln_input_kernel<<<ROWS_TOT, 256, 0, stream>>>(x, lat, gx, bx, gl, bl, kv_in);

  // 3) kv = kv_in @ Wkv  -> bf16 [8*4160, 1024] (k | v halves)
  gemm64_bf16<<<dim3(DIMX / 64, ROWS_TOT / 64, 1), 32, 0, stream>>>(
      kv_in, 0, DIMX, WkvT, kvout, 0, DIMX, DIMX, 0);

  // 4) q = ln(latents) @ Wq  (latent rows are the tail of each batch in kv_in)
  gemm64_bf16<<<dim3(INNERX / 64, 1, BATCH), 32, 0, stream>>>(
      kv_in + (size_t)NSEQ * DIMX, (size_t)NTOT * DIMX, DIMX,
      WqT, qb, (size_t)MLAT * INNERX, INNERX, DIMX, 0);

  // 5) v transpose -> vT [b,h,64,4160]  (overwrites kv_in region; safe in stream order)
  transpose_v_kernel<<<dim3(NTOT / 64, BATCH * HEADS), 256, 0, stream>>>(kvout, vT);

  // 6) flash attention partials: 13 j-chunks x 64 (b,h) = 832 blocks
  attn_partial_kernel<<<dim3(NCHUNK, BATCH * HEADS), 128, 0, stream>>>(
      qb, kvout, vT, msk, pout, pm, pl);

  // 7) merge partials -> aout bf16 [512, 512]
  attn_combine_kernel<<<BATCH * HEADS, 256, 0, stream>>>(pout, pm, pl, aout);

  // 8) out projection -> f32 proj [512, 1024]
  gemm64_bf16<<<dim3(DIMX / 64, (BATCH * MLAT) / 64, 1), 32, 0, stream>>>(
      aout, 0, INNERX, WoT, proj, 0, DIMX, INNERX, 1);

  // 9) final layernorm -> d_out
  ln_out_kernel<<<BATCH * MLAT, 256, 0, stream>>>(proj, go, bo, out);
}
